// CoreQuantumModule_38551626448910
// MI455X (gfx1250) — matrix operand order repaired and checked by reading, not compiled, not run on hardware
//
#include <hip/hip_runtime.h>
#include <hip/hip_bf16.h>

#define NQ      14
#define NSTATE  16384
#define BLOCK   256

typedef __attribute__((ext_vector_type(16))) _Float16 v16h;
typedef __attribute__((ext_vector_type(8)))  float    v8f;

// symmetric 2-amplitude butterfly: new_a = c*a + s*(a_im-part of b rotated by -i)
__device__ __forceinline__ void bfly(float c, float s,
                                     float& ar, float& ai, float& br, float& bi) {
  float nar = fmaf(c, ar,  s * bi);
  float nai = fmaf(c, ai, -s * br);
  float nbr = fmaf(c, br,  s * ai);
  float nbi = fmaf(c, bi, -s * ar);
  ar = nar; ai = nai; br = nbr; bi = nbi;
}

// insert zero bits at positions pLo < pHi into a 12-bit value
__device__ __forceinline__ int expand2(int r, int pLo, int pHi) {
  int lo  = r & ((1 << pLo) - 1);
  int mid = (r >> pLo) & ((1 << (pHi - pLo - 1)) - 1);
  int hi  = r >> (pHi - 1);
  return (hi << (pHi + 1)) | (mid << (pLo + 1)) | lo;
}

// composed MCX permutation: state_final[i] = state_pre[T(i)]
__device__ __forceinline__ int mcx_T(int j) {
#pragma unroll
  for (int w = 13; w >= 0; --w) {
    const int c1 = w, c2 = (w + 1) % NQ, t = (w + 2) % NQ;
    int b1 = (j >> (13 - c1)) & 1;
    int b2 = (j >> (13 - c2)) & 1;
    j ^= (b1 & (b2 ^ 1)) << (13 - t);
  }
  return j;
}

// two commuting butterflies (masks disjoint), 4 amplitudes per quad
__device__ __forceinline__ void quad_pass(float* re, float* im,
                                          int mA, float cA, float sA,
                                          int mB, float cB, float sB,
                                          int pLo, int pHi, int tid) {
  for (int q = tid; q < 4096; q += BLOCK) {
    int i0 = expand2(q, pLo, pHi);
    int i1 = i0 ^ mA, i2 = i0 ^ mB, i3 = i0 ^ mA ^ mB;
    float a0r = re[i0], a0i = im[i0], a1r = re[i1], a1i = im[i1];
    float a2r = re[i2], a2i = im[i2], a3r = re[i3], a3i = im[i3];
    bfly(cA, sA, a0r, a0i, a1r, a1i);
    bfly(cA, sA, a2r, a2i, a3r, a3i);
    bfly(cB, sB, a0r, a0i, a2r, a2i);
    bfly(cB, sB, a1r, a1i, a3r, a3i);
    re[i0] = a0r; im[i0] = a0i; re[i1] = a1r; im[i1] = a1i;
    re[i2] = a2r; im[i2] = a2i; re[i3] = a3r; im[i3] = a3i;
  }
}

__device__ __forceinline__ void pair_pass(float* re, float* im,
                                          int mask, float c, float s,
                                          int pRep, int tid) {
  for (int q = tid; q < 8192; q += BLOCK) {
    int i0 = ((q >> pRep) << (pRep + 1)) | (q & ((1 << pRep) - 1));
    int i1 = i0 ^ mask;
    float ar = re[i0], ai = im[i0], br = re[i1], bi = im[i1];
    bfly(c, s, ar, ai, br, bi);
    re[i0] = ar; im[i0] = ai; re[i1] = br; im[i1] = bi;
  }
}

__global__ void __launch_bounds__(BLOCK)
qcirc_kernel(const float* __restrict__ cp, const float* __restrict__ p,
             float* __restrict__ out) {
  extern __shared__ float smem[];
  float* re = smem;
  float* im = smem + NSTATE;

  __shared__ float sC1[NQ], sS1[NQ], sC2[NQ], sS2[NQ];
  __shared__ float sCxa[7], sSxa[7], sCxb[7], sSxb[7];
  __shared__ float sTheta[NQ];
  __shared__ float sMlo[128], sMhi[128], sPhiLo[128], sPhiHi[128];
  __shared__ float sR[512];       // stage-1 row sums (per i>>5)
  __shared__ float sRed[8 * 32];  // stage-1 cross-wave column partials
  __shared__ float sRed2[32];     // stage-2 column partials
  __shared__ float sG[16];        // stage-2 group sums (qubits 0..3)

  const int tid = threadIdx.x;
  const int b   = blockIdx.x;
  const float* cpb = cp + b * NQ;
  const float* pb  = p  + b * 3 * NQ;

  // ---------------- parameters & trig ----------------
  if (tid < NQ)            { float s, c; sincosf(0.5f * pb[tid], &s, &c);            sC1[tid] = c; sS1[tid] = s; }
  else if (tid < 2 * NQ)   { int w = tid - NQ;     float s, c; sincosf(0.5f * pb[2 * NQ + w], &s, &c); sC2[w] = c; sS2[w] = s; }
  else if (tid < 3 * NQ)   { int w = tid - 2 * NQ; sTheta[w] = pb[NQ + w]; }
  else if (tid < 3 * NQ + 7)  { int ws = tid - 3 * NQ;     float s, c; sincosf(0.5f * cpb[ws],     &s, &c); sCxa[ws] = c; sSxa[ws] = s; }
  else if (tid < 3 * NQ + 14) { int ws = tid - 3 * NQ - 7; float s, c; sincosf(0.5f * cpb[7 + ws], &s, &c); sCxb[ws] = c; sSxb[ws] = s; }
  __syncthreads();

  // factorized tables: bit position pos of i maps to qubit 13-pos
  if (tid < 128) {
    int l = tid; float m = 1.f, ph = 0.f;
#pragma unroll
    for (int q5 = 0; q5 < 7; ++q5) {            // positions 0..6 -> qubits 13..7
      int bit = (l >> q5) & 1;
      m  *= bit ? sS1[13 - q5] : sC1[13 - q5];
      ph += sTheta[13 - q5] * (bit ? 0.5f : -0.5f);
    }
    sMlo[l] = m; sPhiLo[l] = ph;
  } else {
    int h = tid - 128; float m = 1.f, ph = 0.f;
#pragma unroll
    for (int q5 = 0; q5 < 7; ++q5) {            // positions 7..13 -> qubits 6..0
      int bit = (h >> q5) & 1;
      m  *= bit ? sS1[6 - q5] : sC1[6 - q5];
      ph += sTheta[6 - q5] * (bit ? 0.5f : -0.5f);
    }
    sMhi[h] = m; sPhiHi[h] = ph;
  }
  __syncthreads();

  // ---------------- init = |0..0> with RX layer 1 applied analytically ----------------
  // amp(i) = (prod of c/s per bit) * (-i)^popcount(i)
  for (int i = tid; i < NSTATE; i += BLOCK) {
    float m = sMhi[i >> 7] * sMlo[i & 127];
    int k = __popc(i) & 3;
    float vr = (k == 0) ? m : ((k == 2) ? -m : 0.f);
    float vi = (k == 1) ? -m : ((k == 3) ? m : 0.f);
    re[i] = vr; im[i] = vi;
  }
  __syncthreads();

  // ---------------- IsingXX layer A: qubit pairs (0,1)(2,3)...(12,13) ----------------
  quad_pass(re, im, 0x3000, sCxa[0], sSxa[0], 0x0C00, sCxa[1], sSxa[1], 11, 13, tid); __syncthreads();
  quad_pass(re, im, 0x0300, sCxa[2], sSxa[2], 0x00C0, sCxa[3], sSxa[3],  7,  9, tid); __syncthreads();
  quad_pass(re, im, 0x0030, sCxa[4], sSxa[4], 0x000C, sCxa[5], sSxa[5],  3,  5, tid); __syncthreads();
  pair_pass(re, im, 0x0003, sCxa[6], sSxa[6], 1, tid); __syncthreads();

  // ---------------- RZ layer (diagonal): amp *= e^{i phi(i)} ----------------
  for (int i = tid; i < NSTATE; i += BLOCK) {
    float phi = sPhiHi[i >> 7] + sPhiLo[i & 127];
    float sp, cpn; sincosf(phi, &sp, &cpn);
    float r0 = re[i], i0 = im[i];
    re[i] = r0 * cpn - i0 * sp;
    im[i] = r0 * sp + i0 * cpn;
  }
  __syncthreads();

  // ---------------- IsingXX layer B: pairs (1,2)(3,4)...(11,12),(13,0) ----------------
  quad_pass(re, im, 0x1800, sCxb[0], sSxb[0], 0x0600, sCxb[1], sSxb[1], 10, 12, tid); __syncthreads();
  quad_pass(re, im, 0x0180, sCxb[2], sSxb[2], 0x0060, sCxb[3], sSxb[3],  6,  8, tid); __syncthreads();
  quad_pass(re, im, 0x0018, sCxb[4], sSxb[4], 0x0006, sCxb[5], sSxb[5],  2,  4, tid); __syncthreads();
  pair_pass(re, im, 0x2001, sCxb[6], sSxb[6], 13, tid); __syncthreads();

  // ---------------- RX layer 2: 14 single-qubit gates, radix-4 fused ----------------
  quad_pass(re, im, 1 << 13, sC2[0],  sS2[0],  1 << 12, sC2[1],  sS2[1],  12, 13, tid); __syncthreads();
  quad_pass(re, im, 1 << 11, sC2[2],  sS2[2],  1 << 10, sC2[3],  sS2[3],  10, 11, tid); __syncthreads();
  quad_pass(re, im, 1 << 9,  sC2[4],  sS2[4],  1 << 8,  sC2[5],  sS2[5],   8,  9, tid); __syncthreads();
  quad_pass(re, im, 1 << 7,  sC2[6],  sS2[6],  1 << 6,  sC2[7],  sS2[7],   6,  7, tid); __syncthreads();
  quad_pass(re, im, 1 << 5,  sC2[8],  sS2[8],  1 << 4,  sC2[9],  sS2[9],   4,  5, tid); __syncthreads();
  quad_pass(re, im, 1 << 3,  sC2[10], sS2[10], 1 << 2,  sC2[11], sS2[11],  2,  3, tid); __syncthreads();
  quad_pass(re, im, 1 << 1,  sC2[12], sS2[12], 1 << 0,  sC2[13], sS2[13],  0,  1, tid); __syncthreads();

  // ---------------- WMMA epilogue: out[w] = sum_i sign_w(i) * |amp(T(i))|^2 ----------------
  const int lane = tid & 31;
  const int wave = tid >> 5;
  const int Ngrp = lane >> 4;   // 0 or 1 (lane half)
  const int Nn   = lane & 15;   // output column / A row within half

  // B (32x16 f16): columns 0..4 = signs of K bit (4-n), column 5 = ones (row sums).
  // Lane L<16 holds (N=L, K=e<8?e:e+8); lane L>=16 holds (N=L-16, K=(e<8?e:e+8)+8).
  v16h Bv;
#pragma unroll
  for (int e = 0; e < 16; ++e) {
    int K = ((e < 8) ? e : e + 8) + 8 * Ngrp;
    float v = 0.0f;
    if (Nn < 5)       v = 1.0f - 2.0f * (float)((K >> (4 - Nn)) & 1);
    else if (Nn == 5) v = 1.0f;
    Bv[e] = (_Float16)v;
  }

  // Stage 1: i = tile*512 + M*32 + K. Columns 0..4 -> qubits 9..13; col 5 -> R[i>>5].
  float accCol = 0.0f;
  for (int tile = wave; tile < 32; tile += 8) {   // wave-uniform: EXEC stays all-ones
    v16h Ahi, Alo;
#pragma unroll
    for (int e = 0; e < 16; ++e) {
      int v = e >> 1, h = e & 1;
      int K = 2 * v + h + ((v < 4) ? 0 : 8) + 8 * Ngrp;   // documented f16 A layout
      int i = tile * 512 + Nn * 32 + K;                   // M = Nn
      int j = mcx_T(i);
      float rr = re[j], ii = im[j];
      float pv = fmaf(rr, rr, ii * ii);
      _Float16 hh = (_Float16)pv;
      Ahi[e] = hh;
      Alo[e] = (_Float16)(pv - (float)hh);                // hi/lo split ~ f32 accuracy
    }
    v8f Cz = {0.f, 0.f, 0.f, 0.f, 0.f, 0.f, 0.f, 0.f};
    v8f D = __builtin_amdgcn_wmma_f32_16x16x32_f16(false, Alo, false, Bv, (short)0, Cz, false, false);
    D     = __builtin_amdgcn_wmma_f32_16x16x32_f16(false, Ahi, false, Bv, (short)0, D,  false, false);
    if (Nn == 5) {                                        // D[v] = (M=v+8*Ngrp, N=5) row sums
#pragma unroll
      for (int v = 0; v < 8; ++v) sR[tile * 16 + v + 8 * Ngrp] = D[v];
    }
    if (Nn < 5) {
      float s = 0.f;
#pragma unroll
      for (int v = 0; v < 8; ++v) s += D[v];
      accCol += s;
    }
  }
  sRed[wave * 32 + lane] = accCol;
  __syncthreads();

  if (tid < 5) {                                          // qubits 9..13
    float t = 0.f;
#pragma unroll
    for (int w = 0; w < 8; ++w) t += sRed[w * 32 + tid] + sRed[w * 32 + tid + 16];
    out[b * NQ + 9 + tid] = t;
  }

  // Stage 2: r = M2*32 + K2 over the 512 row sums. Columns 0..4 -> qubits 4..8;
  // col-5 row sums -> 16 group sums for qubits 0..3.
  if (wave == 0) {
    v16h A2h, A2l;
#pragma unroll
    for (int e = 0; e < 16; ++e) {
      int v = e >> 1, h = e & 1;
      int K = 2 * v + h + ((v < 4) ? 0 : 8) + 8 * Ngrp;
      float pv = sR[Nn * 32 + K];
      _Float16 hh = (_Float16)pv;
      A2h[e] = hh;
      A2l[e] = (_Float16)(pv - (float)hh);
    }
    v8f Cz2 = {0.f, 0.f, 0.f, 0.f, 0.f, 0.f, 0.f, 0.f};
    v8f D2 = __builtin_amdgcn_wmma_f32_16x16x32_f16(false, A2l, false, Bv, (short)0, Cz2, false, false);
    D2     = __builtin_amdgcn_wmma_f32_16x16x32_f16(false, A2h, false, Bv, (short)0, D2,  false, false);
    if (Nn == 5) {
#pragma unroll
      for (int v = 0; v < 8; ++v) sG[v + 8 * Ngrp] = D2[v];
    }
    float s2 = 0.f;
#pragma unroll
    for (int v = 0; v < 8; ++v) s2 += D2[v];
    sRed2[lane] = s2;
  }
  __syncthreads();

  if (tid < 5) {                                          // qubits 4..8
    out[b * NQ + 4 + tid] = sRed2[tid] + sRed2[tid + 16];
  } else if (tid < 9) {                                   // qubits 0..3
    int w = tid - 5;
    float t = 0.f;
#pragma unroll
    for (int m2 = 0; m2 < 16; ++m2) {
      float sgn = 1.0f - 2.0f * (float)((m2 >> (3 - w)) & 1);
      t += sgn * sG[m2];
    }
    out[b * NQ + w] = t;
  }
}

extern "C" void kernel_launch(void* const* d_in, const int* in_sizes, int n_in,
                              void* d_out, int out_size, void* d_ws, size_t ws_size,
                              hipStream_t stream) {
  const float* cp = (const float*)d_in[0];   // (512, 14) f32
  const float* p  = (const float*)d_in[1];   // (512, 3, 14) f32
  float* out = (float*)d_out;                // (512, 14) f32
  const int B = in_sizes[0] / NQ;            // 512 workgroups, one statevector each
  const size_t shmem = (size_t)2 * NSTATE * sizeof(float);   // 128 KB LDS statevector
  qcirc_kernel<<<dim3(B), dim3(BLOCK), shmem, stream>>>(cp, p, out);
}
